// RHWFNet_48919677501890
// MI455X (gfx1250) — matrix operand order repaired and checked by reading, not compiled, not run on hardware
//
#include <hip/hip_runtime.h>
#include <hip/hip_bf16.h>

typedef __bf16 bf16_t;
typedef __attribute__((ext_vector_type(16))) __bf16 v16bf;
typedef __attribute__((ext_vector_type(8)))  __bf16 v8bf;
typedef __attribute__((ext_vector_type(2)))  __bf16 v2bf;
typedef __attribute__((ext_vector_type(8)))  float  v8f;

#define PLANE 9216               // 96*96
#define CH    128
#define BATCH 4
#define TEN_F ((size_t)CH * PLANE)          // floats per batch image
#define TEN_ALL ((size_t)BATCH * TEN_F)     // floats per tensor

// ---------------------------------------------------------------------------
// fp32 -> bf16 weight conversion
// ---------------------------------------------------------------------------
__global__ void f2bf_kernel(const float* __restrict__ s, bf16_t* __restrict__ d, int n) {
    int i = blockIdx.x * 256 + threadIdx.x;
    if (i < n) d[i] = (bf16_t)s[i];
}

__device__ __forceinline__ unsigned pack_bf16(float lo, float hi) {
    union { v2bf h; unsigned u; } cv;
    cv.h[0] = (bf16_t)lo;
    cv.h[1] = (bf16_t)hi;
    return cv.u;
}

// ---------------------------------------------------------------------------
// Fused 128x128 1x1-conv GEMM over 64-pixel tiles, bf16 WMMA, f32 accum.
// MODE 0: Y = W*X
// MODE 1: Y = relu(W*X + bias)
// MODE 2: Y = LayerNorm(W*X + resid; gamma,beta)
// MODE 3: Y = LayerNorm(W*X + bias; gamma,beta) + resid   (resid may alias Y)
// grid = (PLANE/64, BATCH), block = 256 (8 waves).
// Whole 128(K) x 64(N) B-tile staged to LDS once (bf16, pairs packed to b32);
// row stride 136 bf16 (68 dwords) -> conflict-free ds_load_b128 fragments.
// ---------------------------------------------------------------------------
template<int MODE>
__global__ __launch_bounds__(256)
void gemm128_kernel(const bf16_t* __restrict__ Wt, const float* __restrict__ X,
                    float* Y,
                    const float* __restrict__ bias,
                    const float* resid,
                    const float* __restrict__ gamma,
                    const float* __restrict__ beta)
{
    __shared__ unsigned sBu[64 * 68];    // [pixel][k-pair], 17408 B

    const int tid   = threadIdx.x;
    const int lane  = tid & 31;
    const int wid   = tid >> 5;
    const int b     = blockIdx.y;
    const int p0    = blockIdx.x * 64;
    const size_t xbase = (size_t)b * TEN_F;

    const int lhalf = lane >> 4;            // 0: lanes 0-15, 1: lanes 16-31
    const int m     = wid * 16 + (lane & 15);

    // ---- preload all 4 A fragments (this lane's slices of weight row m)
    v16bf a[4];
#pragma unroll
    for (int ks = 0; ks < 4; ++ks) {
        const bf16_t* ar = Wt + (size_t)m * 128 + ks * 32 + lhalf * 8;
        v8bf alo = *(const v8bf*)(ar);
        v8bf ahi = *(const v8bf*)(ar + 16);
#pragma unroll
        for (int i = 0; i < 8; ++i) { a[ks][i] = alo[i]; a[ks][8 + i] = ahi[i]; }
    }

    // ---- stage full B tile: X[0..128)[p0..p0+64) -> bf16 LDS, transposed.
    // Each thread handles one channel pair (k=2kp, 2kp+1) x 16 pixels:
    // fp32 pair packs into one dword (v_cvt_pk_bf16_f32) -> ds_store_b32.
    {
        const int kp = tid >> 2;            // 0..63
        const int n0 = (tid & 3) * 16;      // 0,16,32,48
        const float* r0 = X + xbase + (size_t)(2 * kp)     * PLANE + p0 + n0;
        const float* r1 = X + xbase + (size_t)(2 * kp + 1) * PLANE + p0 + n0;
#pragma unroll
        for (int i = 0; i < 4; ++i) {
            float4 f0 = ((const float4*)r0)[i];
            float4 f1 = ((const float4*)r1)[i];
            sBu[(n0 + i * 4 + 0) * 68 + kp] = pack_bf16(f0.x, f1.x);
            sBu[(n0 + i * 4 + 1) * 68 + kp] = pack_bf16(f0.y, f1.y);
            sBu[(n0 + i * 4 + 2) * 68 + kp] = pack_bf16(f0.z, f1.z);
            sBu[(n0 + i * 4 + 3) * 68 + kp] = pack_bf16(f0.w, f1.w);
        }
    }
    __syncthreads();

    v8f acc[4];
#pragma unroll
    for (int t = 0; t < 4; ++t)
#pragma unroll
        for (int r = 0; r < 8; ++r) acc[t][r] = 0.0f;

    const bf16_t* sB = (const bf16_t*)sBu;
#pragma unroll
    for (int ks = 0; ks < 4; ++ks) {
#pragma unroll
        for (int t = 0; t < 4; ++t) {
            const bf16_t* br =
                sB + (size_t)(t * 16 + (lane & 15)) * 136 + ks * 32 + lhalf * 8;
            v8bf blo = *(const v8bf*)(br);
            v8bf bhi = *(const v8bf*)(br + 16);
            v16bf bb;
#pragma unroll
            for (int i = 0; i < 8; ++i) { bb[i] = blo[i]; bb[8 + i] = bhi[i]; }
            acc[t] = __builtin_amdgcn_wmma_f32_16x16x32_bf16(
                false, a[ks], false, bb, (short)0, acc[t], false, false);
        }
    }

    const int mrow = wid * 16 + lhalf * 8;  // base output channel of acc rows
    const int ncol = lane & 15;

    if constexpr (MODE <= 1) {
#pragma unroll
        for (int t = 0; t < 4; ++t) {
            int p = p0 + t * 16 + ncol;
#pragma unroll
            for (int r = 0; r < 8; ++r) {
                int o = mrow + r;
                float v = acc[t][r];
                if constexpr (MODE == 1) v = fmaxf(v + bias[o], 0.0f);
                Y[xbase + (size_t)o * PLANE + p] = v;
            }
        }
    } else {
        __shared__ float sAcc[128 * 65];
        __shared__ float redS[256];
        __shared__ float redQ[256];
        __shared__ float muA[64];
        __shared__ float rsA[64];

#pragma unroll
        for (int t = 0; t < 4; ++t)
#pragma unroll
            for (int r = 0; r < 8; ++r)
                sAcc[(mrow + r) * 65 + t * 16 + ncol] = acc[t][r];
        __syncthreads();

        const int pix  = tid & 63;
        const int part = tid >> 6;        // 4 parts x 32 channels
        const int p    = p0 + pix;

        float s = 0.0f, q = 0.0f;
        for (int c = part * 32; c < part * 32 + 32; ++c) {
            float tv = sAcc[c * 65 + pix];
            if constexpr (MODE == 2) tv += resid[xbase + (size_t)c * PLANE + p];
            else                     tv += bias[c];
            sAcc[c * 65 + pix] = tv;
            s += tv; q += tv * tv;
        }
        redS[tid] = s;
        redQ[tid] = q;
        __syncthreads();
        if (part == 0) {
            float S  = redS[pix] + redS[64 + pix] + redS[128 + pix] + redS[192 + pix];
            float Qq = redQ[pix] + redQ[64 + pix] + redQ[128 + pix] + redQ[192 + pix];
            float mu  = S * (1.0f / 128.0f);
            float var = Qq * (1.0f / 128.0f) - mu * mu;
            muA[pix] = mu;
            rsA[pix] = rsqrtf(var + 1e-6f);
        }
        __syncthreads();
        float mu = muA[pix], rs = rsA[pix];
        for (int c = part * 32; c < part * 32 + 32; ++c) {
            float tv = sAcc[c * 65 + pix];
            float yv = gamma[c] * (tv - mu) * rs + beta[c];
            if constexpr (MODE == 3) yv += resid[xbase + (size_t)c * PLANE + p];
            Y[xbase + (size_t)c * PLANE + p] = yv;
        }
    }
}

// ---------------------------------------------------------------------------
// 5x5 local attention, NH=4 heads x 32 ch, softmax temperature sqrt(32).
// Halo tiles staged LDS-direct via global_load_async_to_lds_b32 (ASYNCcnt),
// zero-fill for out-of-image halo lanes. grid = (36, BATCH), block 256.
// O may alias Q (per-pixel read-before-write).
// ---------------------------------------------------------------------------
__device__ __forceinline__ void stage_tile_async(
    const float* __restrict__ G, float* sT, size_t base, int c0,
    int gy0, int gx0, int tid)
{
    for (int idx = tid; idx < 20 * 20 * 32; idx += 256) {
        int c = idx & 31, rem = idx >> 5;
        int xx = rem % 20, yy = rem / 20;
        int gy = gy0 + yy, gx = gx0 + xx;
        float* ldst = &sT[(yy * 20 + xx) * 33 + c];
        if (gy >= 0 && gy < 96 && gx >= 0 && gx < 96) {
            const float* gsrc = G + base + (size_t)(c0 + c) * PLANE + gy * 96 + gx;
            unsigned laddr = (unsigned)(uintptr_t)ldst;   // LDS byte offset (low 32b)
            asm volatile("global_load_async_to_lds_b32 %0, %1, off"
                         :: "v"(laddr), "v"(gsrc) : "memory");
        } else {
            *ldst = 0.0f;
        }
    }
    asm volatile("s_wait_asynccnt 0" ::: "memory");
}

__global__ __launch_bounds__(256)
void attn5_kernel(const float* Q, const float* __restrict__ K,
                  const float* __restrict__ V, float* O)
{
    __shared__ float sT[20 * 20 * 33];   // k/v halo tile, padded channel dim

    const int tid = threadIdx.x;
    const int tx  = tid & 15, ty = tid >> 4;
    const int b   = blockIdx.y;
    const int tY  = blockIdx.x / 6, tX = blockIdx.x % 6;
    const int y   = tY * 16 + ty, x = tX * 16 + tx;
    const size_t base = (size_t)b * TEN_F;
    const int gy0 = tY * 16 - 2, gx0 = tX * 16 - 2;

    for (int h = 0; h < 4; ++h) {
        const int c0 = h * 32;

        // ---- stage K tile with halo (async to LDS, zero-padded)
        stage_tile_async(K, sT, base, c0, gy0, gx0, tid);
        __syncthreads();

        // prefetch this head's V plane near our pixel while we do correlations
        __builtin_prefetch(V + base + (size_t)c0 * PLANE + y * 96 + x, 0, 1);

        float qr[32];
#pragma unroll
        for (int c = 0; c < 32; ++c)
            qr[c] = Q[base + (size_t)(c0 + c) * PLANE + y * 96 + x];

        float w[25];
        float mx = -1e30f;
#pragma unroll
        for (int j = 0; j < 25; ++j) {
            int dy = j / 5, dx = j % 5;
            const float* row = sT + ((ty + dy) * 20 + tx + dx) * 33;
            float s = 0.0f;
#pragma unroll
            for (int c = 0; c < 32; ++c) s += qr[c] * row[c];
            s *= 0.17677669529663687f;          // 1/sqrt(32)
            w[j] = s;
            mx = fmaxf(mx, s);
        }
        float sum = 0.0f;
#pragma unroll
        for (int j = 0; j < 25; ++j) { w[j] = __expf(w[j] - mx); sum += w[j]; }
        float inv = 1.0f / sum;
#pragma unroll
        for (int j = 0; j < 25; ++j) w[j] *= inv;
        __syncthreads();

        // ---- stage V tile (reuse sT)
        stage_tile_async(V, sT, base, c0, gy0, gx0, tid);
        __syncthreads();

        float acc[32];
#pragma unroll
        for (int c = 0; c < 32; ++c) acc[c] = 0.0f;
#pragma unroll
        for (int j = 0; j < 25; ++j) {
            int dy = j / 5, dx = j % 5;
            float aw = w[j];
            const float* row = sT + ((ty + dy) * 20 + tx + dx) * 33;
#pragma unroll
            for (int c = 0; c < 32; ++c) acc[c] += aw * row[c];
        }
#pragma unroll
        for (int c = 0; c < 32; ++c)
            O[base + (size_t)(c0 + c) * PLANE + y * 96 + x] = acc[c];
        __syncthreads();
    }
}

// ---------------------------------------------------------------------------
// Host orchestration
// ---------------------------------------------------------------------------
extern "C" void kernel_launch(void* const* d_in, const int* in_sizes, int n_in,
                              void* d_out, int out_size, void* d_ws, size_t ws_size,
                              hipStream_t stream)
{
    (void)in_sizes; (void)n_in; (void)out_size; (void)ws_size;

    const float* in1   = (const float*)d_in[0];
    const float* in2   = (const float*)d_in[1];
    const float* wsrc[10] = {
        (const float*)d_in[2],  // wq_s
        (const float*)d_in[3],  // wk_s
        (const float*)d_in[4],  // wv_s
        (const float*)d_in[5],  // fc_s
        (const float*)d_in[8],  // wq_c
        (const float*)d_in[9],  // wk_c
        (const float*)d_in[10], // wv_c
        (const float*)d_in[11], // fc_c
        (const float*)d_in[14], // w1
        (const float*)d_in[16], // w2
    };
    const float* lnw_s = (const float*)d_in[6];
    const float* lnb_s = (const float*)d_in[7];
    const float* lnw_c = (const float*)d_in[12];
    const float* lnb_c = (const float*)d_in[13];
    const float* b1    = (const float*)d_in[15];
    const float* b2    = (const float*)d_in[17];
    const float* lnw_f = (const float*)d_in[18];
    const float* lnb_f = (const float*)d_in[19];

    char*   ws   = (char*)d_ws;
    bf16_t* wbf  = (bf16_t*)ws;
    auto W = [&](int i) { return wbf + (size_t)i * 16384; };
    float* fbase = (float*)(ws + 393216);      // after bf16 weights (aligned)

    float* slf1 = fbase;
    float* slf2 = fbase + TEN_ALL;
    float* qb   = fbase + 2 * TEN_ALL;
    float* kb   = fbase + 3 * TEN_ALL;
    float* vb   = fbase + 4 * TEN_ALL;
    float* out1 = (float*)d_out;               // crs1 then FFN(crs1) in place
    float* out2 = out1 + TEN_ALL;              // crs2 then FFN(crs2) in place

    for (int i = 0; i < 10; ++i)
        f2bf_kernel<<<64, 256, 0, stream>>>(wsrc[i], W(i), 16384);

    dim3 gg(PLANE / 64, BATCH);
    dim3 ga(36, BATCH);

    auto mha = [&](const float* xq, const float* xkv,
                   int wq, int wk, int wv, int wfc,
                   const float* g, const float* bt, float* out) {
        gemm128_kernel<0><<<gg, 256, 0, stream>>>(W(wq),  xq,  qb, nullptr, nullptr, nullptr, nullptr);
        gemm128_kernel<0><<<gg, 256, 0, stream>>>(W(wk),  xkv, kb, nullptr, nullptr, nullptr, nullptr);
        gemm128_kernel<0><<<gg, 256, 0, stream>>>(W(wv),  xkv, vb, nullptr, nullptr, nullptr, nullptr);
        attn5_kernel<<<ga, 256, 0, stream>>>(qb, kb, vb, qb);   // att output aliases qb (safe)
        gemm128_kernel<2><<<gg, 256, 0, stream>>>(W(wfc), qb, out, nullptr, xq, g, bt);
    };

    mha(in1,  in1,  0, 1, 2, 3, lnw_s, lnb_s, slf1);
    mha(in2,  in2,  0, 1, 2, 3, lnw_s, lnb_s, slf2);
    mha(slf1, slf2, 4, 5, 6, 7, lnw_c, lnb_c, out1);   // crs1
    mha(slf2, slf1, 4, 5, 6, 7, lnw_c, lnb_c, out2);   // crs2

    auto ffn = [&](float* xio) {
        gemm128_kernel<1><<<gg, 256, 0, stream>>>(W(8), xio, kb, b1, nullptr, nullptr, nullptr);
        gemm128_kernel<3><<<gg, 256, 0, stream>>>(W(9), kb, xio, b2, xio, lnw_f, lnb_f);
    };
    ffn(out1);
    ffn(out2);
}
